// PointNet2Backbone_20220706029882
// MI455X (gfx1250) — matrix operand order repaired and checked by reading, not compiled, not run on hardware
//
#include <hip/hip_runtime.h>
#include <stdint.h>

// ---------------------------------------------------------------------------
// PointNet++ backbone for MI455X (gfx1250, wave32).
// GEMMs run on v_wmma_f32_16x16x32_bf16 (bf16 in, f32 accumulate), software
// pipelined 2-deep, one 16x32 output tile per wave (A-fragment reuse).
// ---------------------------------------------------------------------------

typedef __bf16 bf16;
typedef __attribute__((ext_vector_type(16))) __bf16 v16bf;
typedef __attribute__((ext_vector_type(8)))  float  v8f;
typedef __attribute__((ext_vector_type(4)))  unsigned int u32x4;
typedef __attribute__((ext_vector_type(4)))  int i32x4;
typedef __attribute__((ext_vector_type(8)))  int i32x8;

static constexpr int B_ = 4, N_ = 8192, C_OUT_ = 512;
static constexpr float BN_EPS_ = 1e-5f;

// ---------------------------------------------------------------------------
// Elementwise converts
// ---------------------------------------------------------------------------
__global__ void k_f2bf(const float* __restrict__ s, bf16* __restrict__ d, int tot) {
  int t = blockIdx.x * blockDim.x + threadIdx.x;
  if (t < tot) d[t] = (bf16)s[t];
}

// weight f32 [cout,cin] -> bf16 [cout,kpad] (dst pre-zeroed)
__global__ void k_w2bf(const float* __restrict__ s, bf16* __restrict__ d,
                       int cin, int kp, int tot) {
  int t = blockIdx.x * blockDim.x + threadIdx.x;
  if (t >= tot) return;
  int r = t / cin, c = t % cin;
  d[(size_t)r * kp + c] = (bf16)s[t];
}

__global__ void k_copy_bf16_cols(const bf16* __restrict__ s, bf16* __restrict__ d,
                                 int cols, int lds_, int ldd, int coloff, int tot) {
  int t = blockIdx.x * blockDim.x + threadIdx.x;
  if (t >= tot) return;
  int r = t / cols, c = t % cols;
  d[(size_t)r * ldd + coloff + c] = s[(size_t)r * lds_ + c];
}

__global__ void k_f32cols_bf16(const float* __restrict__ s, bf16* __restrict__ d,
                               int cols, int ldd, int coloff, int tot) {
  int t = blockIdx.x * blockDim.x + threadIdx.x;
  if (t >= tot) return;
  int r = t / cols, c = t % cols;
  d[(size_t)r * ldd + coloff + c] = (bf16)s[(size_t)r * cols + c];
}

// ---------------------------------------------------------------------------
// Farthest point sampling: one block per batch, sequential scan like reference
// ---------------------------------------------------------------------------
__global__ __launch_bounds__(256)
void k_fps(const float* __restrict__ xyz, float* __restrict__ dist,
           int* __restrict__ idx, int n, int npoint) {
  int b = blockIdx.x;
  const float* p = xyz + (size_t)b * n * 3;
  float* d = dist + (size_t)b * n;
  int* id = idx + (size_t)b * npoint;
  __shared__ float s_val[256];
  __shared__ int   s_idx[256];
  __shared__ float s_far[3];
  __shared__ int   s_cur;
  int t = threadIdx.x;
  for (int i = t; i < n; i += 256) d[i] = 1e10f;
  if (t == 0) s_cur = 0;
  __syncthreads();
  for (int s = 0; s < npoint; ++s) {
    int cur = s_cur;
    if (t == 0) {
      id[s] = cur;
      s_far[0] = p[cur * 3 + 0];
      s_far[1] = p[cur * 3 + 1];
      s_far[2] = p[cur * 3 + 2];
    }
    __syncthreads();
    float fx = s_far[0], fy = s_far[1], fz = s_far[2];
    float best = -1.0f; int bi = 0;
    for (int i = t; i < n; i += 256) {
      float dx = p[i * 3 + 0] - fx;
      float dy = p[i * 3 + 1] - fy;
      float dz = p[i * 3 + 2] - fz;
      float dd = dx * dx + dy * dy + dz * dz;
      float m = fminf(d[i], dd);
      d[i] = m;
      if (m > best) { best = m; bi = i; }
    }
    s_val[t] = best; s_idx[t] = bi;
    __syncthreads();
    for (int o = 128; o > 0; o >>= 1) {
      if (t < o && s_val[t + o] > s_val[t]) { s_val[t] = s_val[t + o]; s_idx[t] = s_idx[t + o]; }
      __syncthreads();
    }
    if (t == 0) s_cur = s_idx[0];
    __syncthreads();
  }
}

__global__ void k_gather_xyz(const float* __restrict__ xyz, const int* __restrict__ idx,
                             float* __restrict__ out, int n, int np, int tot) {
  int g = blockIdx.x * blockDim.x + threadIdx.x;
  if (g >= tot) return;
  int b = g / np;
  int s = idx[g];
  const float* p = xyz + ((size_t)b * n + s) * 3;
  out[(size_t)g * 3 + 0] = p[0];
  out[(size_t)g * 3 + 1] = p[1];
  out[(size_t)g * 3 + 2] = p[2];
}

// ball query: smallest-index semantics (matches the reference's sort-then-take)
__global__ void k_ball_query(const float* __restrict__ xyz, const float* __restrict__ nxyz,
                             int* __restrict__ idx, int n, int np, int ns, float r2, int tot) {
  int g = blockIdx.x * blockDim.x + threadIdx.x;
  if (g >= tot) return;
  int b = g / np;
  const float* p = xyz + (size_t)b * n * 3;
  float qx = nxyz[(size_t)g * 3 + 0];
  float qy = nxyz[(size_t)g * 3 + 1];
  float qz = nxyz[(size_t)g * 3 + 2];
  int* out = idx + (size_t)g * ns;
  int cnt = 0, first = 0; bool has = false;
  for (int i = 0; i < n && cnt < ns; ++i) {
    float dx = p[i * 3 + 0] - qx;
    float dy = p[i * 3 + 1] - qy;
    float dz = p[i * 3 + 2] - qz;
    if (dx * dx + dy * dy + dz * dz < r2) {
      if (!has) { first = i; has = true; }
      out[cnt++] = i;
    }
  }
  for (; cnt < ns; ++cnt) out[cnt] = first;
}

// grouped input: row=((b*np+j)*ns+s), cols [0,3)=centered xyz, [3,3+cf)=feats
__global__ void k_group(const float* __restrict__ xyz, const float* __restrict__ nxyz,
                        const bf16* __restrict__ feats, const int* __restrict__ idx,
                        bf16* __restrict__ out, int n, int np, int ns, int cf,
                        int kp, int tot) {
  int g = blockIdx.x * blockDim.x + threadIdx.x;
  if (g >= tot) return;
  int cw = 3 + cf;
  int row = g / cw, c = g % cw;
  int pj = row / ns;
  int b = pj / np;
  int src = idx[row];
  bf16 v;
  if (c < 3)
    v = (bf16)(xyz[((size_t)b * n + src) * 3 + c] - nxyz[(size_t)pj * 3 + c]);
  else
    v = feats[((size_t)b * n + src) * cf + (c - 3)];
  out[(size_t)row * kp + c] = v;
}

// 3-NN inverse-distance interpolation of known feats into dst cols [0,ck)
__global__ void k_three_interp(const float* __restrict__ ux, const float* __restrict__ kx,
                               const bf16* __restrict__ kf, bf16* __restrict__ dst,
                               int nu, int nk, int ck, int ldd, int tot) {
  int g = blockIdx.x * blockDim.x + threadIdx.x;
  if (g >= tot) return;
  int b = g / nu;
  float qx = ux[(size_t)g * 3 + 0];
  float qy = ux[(size_t)g * 3 + 1];
  float qz = ux[(size_t)g * 3 + 2];
  const float* kp_ = kx + (size_t)b * nk * 3;
  float d0 = 1e30f, d1 = 1e30f, d2 = 1e30f; int i0 = 0, i1 = 0, i2 = 0;
  for (int i = 0; i < nk; ++i) {
    float dx = kp_[i * 3 + 0] - qx;
    float dy = kp_[i * 3 + 1] - qy;
    float dz = kp_[i * 3 + 2] - qz;
    float dd = dx * dx + dy * dy + dz * dz;
    if (dd < d0)      { d2 = d1; i2 = i1; d1 = d0; i1 = i0; d0 = dd; i0 = i; }
    else if (dd < d1) { d2 = d1; i2 = i1; d1 = dd; i1 = i; }
    else if (dd < d2) { d2 = dd; i2 = i; }
  }
  float w0 = 1.f / (d0 + 1e-8f), w1 = 1.f / (d1 + 1e-8f), w2 = 1.f / (d2 + 1e-8f);
  float ws = w0 + w1 + w2; w0 /= ws; w1 /= ws; w2 /= ws;
  const bf16* f0 = kf + ((size_t)b * nk + i0) * ck;
  const bf16* f1 = kf + ((size_t)b * nk + i1) * ck;
  const bf16* f2 = kf + ((size_t)b * nk + i2) * ck;
  bf16* o = dst + (size_t)g * ldd;
  for (int c = 0; c < ck; ++c)
    o[c] = (bf16)(w0 * (float)f0[c] + w1 * (float)f1[c] + w2 * (float)f2[c]);
}

// ---------------------------------------------------------------------------
// WMMA GEMM: C[M,Nc] = A[M,K](bf16,row-major,ld=K) * W[Nc,K]^T(bf16,ld=K)
// K multiple of 32 (zero padded); M,Nc multiples of 64 here.
// One 16x32 output tile per wave (A fragment reused over two B tiles),
// 2-deep software pipeline with even/odd fragment rotation (no reg copies).
// Fragment k order: contiguous k = 16*half + s for A and B (not the ISA table order).
// ---------------------------------------------------------------------------
struct Frag { v16bf a, b0, b1; };

__device__ __forceinline__
void load_frag(Frag& f, const bf16* __restrict__ ar,
               const bf16* __restrict__ wr0, const bf16* __restrict__ wr1) {
#pragma unroll
  for (int j = 0; j < 8; ++j) {
    int ka = 2 * j;
    f.a[2 * j]      = ar[ka];
    f.a[2 * j + 1]  = ar[ka + 1];
    f.b0[2 * j]     = wr0[2 * j];
    f.b0[2 * j + 1] = wr0[2 * j + 1];
    f.b1[2 * j]     = wr1[2 * j];
    f.b1[2 * j + 1] = wr1[2 * j + 1];
  }
}

__global__ __launch_bounds__(256)
void k_gemm_bf16(const bf16* __restrict__ A, const bf16* __restrict__ W,
                 float* __restrict__ C, int M, int Nc, int K) {
  const int lane = threadIdx.x & 31;
  const int wv   = threadIdx.x >> 5;
  const int row0 = blockIdx.x * 64 + (wv & 3) * 16;   // 64 rows / block
  const int col0 = blockIdx.y * 64 + (wv >> 2) * 32;  // 64 cols / block (2 tiles/wave)
  if (row0 >= M || col0 >= Nc) return;                 // wave-uniform
  const int half = lane >> 4;
  const int sub  = lane & 15;
  // per-lane base pointers (A lane: K in {0..15}, +16 for hi half;
  //                          B lane: K in {0..15}, +16 for hi half)
  const bf16* aptr  = A + (size_t)(row0 + sub) * K + half * 16;
  const bf16* wptr0 = W + (size_t)(col0 + sub) * K + half * 16;
  const bf16* wptr1 = wptr0 + (size_t)16 * K;
  v8f acc0 = {0.f, 0.f, 0.f, 0.f, 0.f, 0.f, 0.f, 0.f};
  v8f acc1 = acc0;

  Frag f0, f1;
  load_frag(f0, aptr, wptr0, wptr1);
  int k = 0;
  for (;;) {
    int kn = k + 32;
    if (kn < K) {
      load_frag(f1, aptr + kn, wptr0 + kn, wptr1 + kn);  // issue next loads
      __builtin_prefetch(aptr + kn + 32, 0, 1);          // global_prefetch_b8
      __builtin_prefetch(wptr0 + kn + 32, 0, 1);
    }
    acc0 = __builtin_amdgcn_wmma_f32_16x16x32_bf16(false, f0.a, false, f0.b0,
                                                   (short)0, acc0, false, false);
    acc1 = __builtin_amdgcn_wmma_f32_16x16x32_bf16(false, f0.a, false, f0.b1,
                                                   (short)0, acc1, false, false);
    k = kn;
    if (k >= K) break;
    kn = k + 32;
    if (kn < K) load_frag(f0, aptr + kn, wptr0 + kn, wptr1 + kn);
    acc0 = __builtin_amdgcn_wmma_f32_16x16x32_bf16(false, f1.a, false, f1.b0,
                                                   (short)0, acc0, false, false);
    acc1 = __builtin_amdgcn_wmma_f32_16x16x32_bf16(false, f1.a, false, f1.b1,
                                                   (short)0, acc1, false, false);
    k = kn;
    if (k >= K) break;
  }
#pragma unroll
  for (int v = 0; v < 8; ++v) {              // D: VGPR v -> row v (+8 hi half), col=lane&15
    int r = row0 + half * 8 + v;
    C[(size_t)r * Nc + col0 + sub]      = acc0[v];
    C[(size_t)r * Nc + col0 + 16 + sub] = acc1[v];
  }
}

// ---------------------------------------------------------------------------
// Batch-norm stats (per output channel over all rows) + fused normalizers
// ---------------------------------------------------------------------------
__global__ __launch_bounds__(256)
void k_bn_stats(const float* __restrict__ x, float* __restrict__ mv, int rows, int cols) {
  int c = blockIdx.x;
  float s = 0.f, ss = 0.f;
  for (int r = threadIdx.x; r < rows; r += 256) {
    float v = x[(size_t)r * cols + c];
    s += v; ss += v * v;
  }
  __shared__ float sh0[256], sh1[256];
  sh0[threadIdx.x] = s; sh1[threadIdx.x] = ss;
  __syncthreads();
  for (int o = 128; o > 0; o >>= 1) {
    if (threadIdx.x < o) {
      sh0[threadIdx.x] += sh0[threadIdx.x + o];
      sh1[threadIdx.x] += sh1[threadIdx.x + o];
    }
    __syncthreads();
  }
  if (threadIdx.x == 0) {
    float m = sh0[0] / (float)rows;
    float v = sh1[0] / (float)rows - m * m;
    mv[2 * c] = m; mv[2 * c + 1] = v;
  }
}

__global__ void k_bn_norm_bf16(const float* __restrict__ x, const float* __restrict__ mv,
                               const float* __restrict__ gg, const float* __restrict__ bb,
                               bf16* __restrict__ dst, int cols, int ldd, int coloff, int tot) {
  int t = blockIdx.x * blockDim.x + threadIdx.x;
  if (t >= tot) return;
  int r = t / cols, c = t % cols;
  float m = mv[2 * c], v = mv[2 * c + 1];
  float y = (x[(size_t)r * cols + c] - m) * rsqrtf(v + BN_EPS_) * gg[c] + bb[c];
  dst[(size_t)r * ldd + coloff + c] = (bf16)fmaxf(y, 0.f);
}

// BN + ReLU + max over ns neighbors, for SA layer-2 outputs
__global__ void k_bn_norm_maxpool(const float* __restrict__ x, const float* __restrict__ mv,
                                  const float* __restrict__ gg, const float* __restrict__ bb,
                                  bf16* __restrict__ dst, int ns, int cols, int ldd,
                                  int coloff, int tot) {
  int t = blockIdx.x * blockDim.x + threadIdx.x;
  if (t >= tot) return;
  int r = t / cols, c = t % cols;
  float m = mv[2 * c];
  float rs = rsqrtf(mv[2 * c + 1] + BN_EPS_);
  float g_ = gg[c], b_ = bb[c];
  float best = -1e30f;
  const float* col = x + c;
  for (int s = 0; s < ns; ++s) {
    float y = (col[((size_t)r * ns + s) * cols] - m) * rs * g_ + b_;
    y = fmaxf(y, 0.f);
    best = fmaxf(best, y);
  }
  dst[(size_t)r * ldd + coloff + c] = (bf16)best;
}

__global__ void k_bn_norm_f32(const float* __restrict__ x, const float* __restrict__ mv,
                              const float* __restrict__ gg, const float* __restrict__ bb,
                              float* __restrict__ dst, int cols, int tot) {
  int t = blockIdx.x * blockDim.x + threadIdx.x;
  if (t >= tot) return;
  int c = t % cols;
  dst[t] = (x[t] - mv[2 * c]) * rsqrtf(mv[2 * c + 1] + BN_EPS_) * gg[c] + bb[c];
}

// ---------------------------------------------------------------------------
// TDM L2-warm: real tensor_load_to_lds of a weight tile + s_wait_tensorcnt.
// Descriptor built per CDNA5 ISA ch.8 (group0: count/lds/global/type,
// group1: data_size=2B, tensor dims/strides, tile dims).
// ---------------------------------------------------------------------------
__global__ void k_tdm_warm(const bf16* __restrict__ w, int rows, int cols) {
#if __has_builtin(__builtin_amdgcn_tensor_load_to_lds)
  __shared__ char lds_buf[16384];
  lds_buf[threadIdx.x] = 0;   // keep LDS allocation alive
  __syncthreads();
  if (threadIdx.x == 0) {
    unsigned long long ga = (unsigned long long)(uintptr_t)w;
    int rowbytes = cols * 2;
    int trows = 16384 / rowbytes;
    if (trows < 1) trows = 1;
    if (trows > rows) trows = rows;
    u32x4 g0;
    g0[0] = 1u;                                               // count=1
    g0[1] = 0u;                                               // lds_addr = base of lds_buf alloc
    g0[2] = (unsigned)(ga & 0xffffffffull);                   // global_addr[31:0]
    g0[3] = (unsigned)((ga >> 32) & 0x1ffffffull) | 0x80000000u; // addr[56:32] | type=2
    i32x8 g1;
    g1[0] = (int)(1u << 16);                                  // data_size = 2 bytes
    g1[1] = (int)(((unsigned)cols & 0xffffu) << 16);          // tensor_dim0 lo
    g1[2] = (int)((((unsigned)cols >> 16) & 0xffffu) | (((unsigned)rows & 0xffffu) << 16));
    g1[3] = (int)((((unsigned)rows >> 16) & 0xffffu) | (((unsigned)cols & 0xffffu) << 16)); // tile_dim0
    g1[4] = (int)((unsigned)trows & 0xffffu);                 // tile_dim1 (tile_dim2=0)
    g1[5] = cols;                                             // tensor_dim0_stride lo
    g1[6] = 0;
    g1[7] = 0;
    i32x4 z4 = {0, 0, 0, 0};
#if defined(__clang_major__) && (__clang_major__ >= 23)
    i32x8 z8 = {0, 0, 0, 0, 0, 0, 0, 0};
    __builtin_amdgcn_tensor_load_to_lds(g0, g1, z4, z4, z8, 0);
#else
    __builtin_amdgcn_tensor_load_to_lds(g0, g1, z4, z4, 0);
#endif
  }
  __builtin_amdgcn_s_wait_tensorcnt(0);
#endif
}

// ---------------------------------------------------------------------------
// Host orchestration
// ---------------------------------------------------------------------------
extern "C" void kernel_launch(void* const* d_in, const int* in_sizes, int n_in,
                              void* d_out, int out_size, void* d_ws, size_t ws_size,
                              hipStream_t stream) {
  (void)in_sizes; (void)n_in; (void)out_size; (void)ws_size;

  const float* xyz0  = (const float*)d_in[0];
  const float* feat0 = (const float*)d_in[1];
  int cur = 2;
  auto nxt = [&]() { return (const float*)d_in[cur++]; };

  struct LayerP { const float *w, *g, *b; int cin, cout; };
  static const int sa_dims[4][3] = {{6,64,128},{259,128,256},{515,256,512},{1027,512,512}};
  static const int fp_dims[4][3] = {{2048,512,512},{1024,512,512},{768,256,256},{259,256,256}};
  static const float radii[4][2] = {{0.02f,0.04f},{0.04f,0.08f},{0.08f,0.16f},{0.16f,0.32f}};
  static const int npL[5] = {N_, 1024, 256, 64, 16};

  // params assumed flattened in setup_inputs() insertion order:
  // sa1..sa4 (2 scales x 2 layers x {w,g,b}), fp4,fp3,fp2,fp1 (2 layers x {w,g,b}),
  // final {w,bias,g,b}
  LayerP sa[4][2][2], fp[4][2], fin;
  for (int i = 0; i < 4; ++i)
    for (int s = 0; s < 2; ++s)
      for (int l = 0; l < 2; ++l) {
        LayerP& L = sa[i][s][l];
        L.w = nxt(); L.g = nxt(); L.b = nxt();
        L.cin = sa_dims[i][l]; L.cout = sa_dims[i][l + 1];
      }
  for (int i = 0; i < 4; ++i)
    for (int l = 0; l < 2; ++l) {
      LayerP& L = fp[i][l];
      L.w = nxt(); L.g = nxt(); L.b = nxt();
      L.cin = fp_dims[i][l]; L.cout = fp_dims[i][l + 1];
    }
  fin.w = nxt();
  const float* fin_bias = nxt(); (void)fin_bias;  // cancels under the final BN
  fin.g = nxt(); fin.b = nxt();
  fin.cin = 256; fin.cout = C_OUT_;

  // -------------------- workspace carve-out --------------------
  size_t off = 0;
  auto alloc = [&](size_t bytes) -> void* {
    off = (off + 255) & ~(size_t)255;
    void* p = (char*)d_ws + off;
    off += bytes;
    return p;
  };
  auto rup = [](int x, int m) { return (x + m - 1) / m * m; };

  float* fps_dist = (float*)alloc((size_t)B_ * N_ * 4);
  int*   fps_idx  = (int*)alloc((size_t)B_ * 1024 * 4);
  int*   ball_idx = (int*)alloc((size_t)B_ * 1024 * 32 * 4);
  float* mv       = (float*)alloc(1024 * 2 * 4);
  float* lx[5]; lx[0] = (float*)xyz0;
  for (int i = 1; i <= 4; ++i) lx[i] = (float*)alloc((size_t)B_ * npL[i] * 3 * 4);
  bf16* l0f = (bf16*)alloc((size_t)B_ * N_ * 3 * 2);
  static const int safC[5] = {0, 256, 512, 1024, 1024};
  bf16* saf[5] = {nullptr, nullptr, nullptr, nullptr, nullptr};
  for (int i = 1; i <= 4; ++i) saf[i] = (bf16*)alloc((size_t)B_ * npL[i] * safC[i] * 2);
  bf16* fpf[4];                                    // fp4,fp3,fp2,fp1 outputs
  fpf[0] = (bf16*)alloc((size_t)B_ * npL[3] * 512 * 2);
  fpf[1] = (bf16*)alloc((size_t)B_ * npL[2] * 512 * 2);
  fpf[2] = (bf16*)alloc((size_t)B_ * npL[1] * 256 * 2);
  fpf[3] = (bf16*)alloc((size_t)B_ * npL[0] * 256 * 2);
  bf16*  bufIn  = (bf16*)alloc((size_t)32768 * 288 * 2);   // largest K-padded input
  bf16*  bufHid = (bf16*)alloc((size_t)131072 * 64 * 2);   // == 32768*256 elems
  float* bufOut = (float*)alloc((size_t)131072 * 128 * 4); // == 32768*512 elems

  bf16* sawb[4][2][2]; bf16* fpwb[4][2]; bf16* finwb;
  for (int i = 0; i < 4; ++i)
    for (int s = 0; s < 2; ++s)
      for (int l = 0; l < 2; ++l)
        sawb[i][s][l] = (bf16*)alloc((size_t)sa[i][s][l].cout * rup(sa[i][s][l].cin, 32) * 2);
  for (int i = 0; i < 4; ++i)
    for (int l = 0; l < 2; ++l)
      fpwb[i][l] = (bf16*)alloc((size_t)fp[i][l].cout * rup(fp[i][l].cin, 32) * 2);
  finwb = (bf16*)alloc((size_t)fin.cout * 256 * 2);

  size_t used = (off + 255) & ~(size_t)255;
  hipMemsetAsync(d_ws, 0, used, stream);   // deterministic zero (also K-padding)

  // -------------------- weight conversion f32 -> padded bf16 --------------------
  auto conv_w = [&](const LayerP& L, bf16* dst) {
    int tot = L.cout * L.cin;
    k_w2bf<<<(tot + 255) / 256, 256, 0, stream>>>(L.w, dst, L.cin, rup(L.cin, 32), tot);
  };
  for (int i = 0; i < 4; ++i)
    for (int s = 0; s < 2; ++s)
      for (int l = 0; l < 2; ++l) conv_w(sa[i][s][l], sawb[i][s][l]);
  for (int i = 0; i < 4; ++i)
    for (int l = 0; l < 2; ++l) conv_w(fp[i][l], fpwb[i][l]);
  conv_w(fin, finwb);

  { int tot = B_ * N_ * 3;
    k_f2bf<<<(tot + 255) / 256, 256, 0, stream>>>(feat0, l0f, tot); }

  // TDM warm of the biggest weight matrix (fp4 layer 1) into L2
  k_tdm_warm<<<1, 32, 0, stream>>>(fpwb[0][0], fp[0][0].cout, rup(fp[0][0].cin, 32));

  // -------------------- common helpers --------------------
  auto gemm = [&](const bf16* A, const bf16* W, float* Cc, int M, int Nc, int K) {
    dim3 g((unsigned)((M + 63) / 64), (unsigned)((Nc + 63) / 64));
    k_gemm_bf16<<<g, 256, 0, stream>>>(A, W, Cc, M, Nc, K);
  };
  auto layer_bf = [&](const bf16* in, int rows, const LayerP& L, const bf16* wb,
                      int kp, bf16* dst, int ldd, int coloff) {
    gemm(in, wb, bufOut, rows, L.cout, kp);
    k_bn_stats<<<L.cout, 256, 0, stream>>>(bufOut, mv, rows, L.cout);
    int tot = rows * L.cout;
    k_bn_norm_bf16<<<(tot + 255) / 256, 256, 0, stream>>>(bufOut, mv, L.g, L.b,
                                                          dst, L.cout, ldd, coloff, tot);
  };

  // -------------------- set-abstraction stages --------------------
  auto run_sa = [&](int i, const float* sxyz, const bf16* sfeat, int cf, int n) {
    int np = npL[i + 1], ns = 32;
    k_fps<<<B_, 256, 0, stream>>>(sxyz, fps_dist, fps_idx, n, np);
    k_gather_xyz<<<(B_ * np + 255) / 256, 256, 0, stream>>>(sxyz, fps_idx, lx[i + 1],
                                                            n, np, B_ * np);
    int c2 = sa_dims[i][2];
    for (int s = 0; s < 2; ++s) {
      float r = radii[i][s];
      int rows = B_ * np * ns;
      int kp = rup(3 + cf, 32);
      hipMemsetAsync(bufIn, 0, (size_t)rows * kp * 2, stream);
      k_ball_query<<<(B_ * np + 255) / 256, 256, 0, stream>>>(sxyz, lx[i + 1], ball_idx,
                                                              n, np, ns, r * r, B_ * np);
      int tot = rows * (3 + cf);
      k_group<<<(tot + 255) / 256, 256, 0, stream>>>(sxyz, lx[i + 1], sfeat, ball_idx,
                                                     bufIn, n, np, ns, cf, kp, tot);
      // layer 1: GEMM + BN + ReLU -> bufHid
      layer_bf(bufIn, rows, sa[i][s][0], sawb[i][s][0], kp, bufHid, sa_dims[i][1], 0);
      // layer 2: GEMM + BN + ReLU + maxpool(ns) -> saf[i+1] at column s*c2
      const LayerP& L2 = sa[i][s][1];
      gemm(bufHid, sawb[i][s][1], bufOut, rows, L2.cout, rup(L2.cin, 32));
      k_bn_stats<<<L2.cout, 256, 0, stream>>>(bufOut, mv, rows, L2.cout);
      int tot2 = B_ * np * L2.cout;
      k_bn_norm_maxpool<<<(tot2 + 255) / 256, 256, 0, stream>>>(bufOut, mv, L2.g, L2.b,
                                                                saf[i + 1], ns, L2.cout,
                                                                2 * c2, s * c2, tot2);
    }
  };
  run_sa(0, lx[0], l0f,    3,    N_);
  run_sa(1, lx[1], saf[1], 256,  1024);
  run_sa(2, lx[2], saf[2], 512,  256);
  run_sa(3, lx[3], saf[3], 1024, 64);

  // -------------------- feature-propagation stages --------------------
  auto run_fp = [&](int m, const float* uxyz, const float* kxyz, int nu, int nk,
                    const bf16* kfeat, int ck, const bf16* ufeat, int cu, bf16* outb) {
    int rows = B_ * nu;
    int kp = rup(ck + cu, 32);
    hipMemsetAsync(bufIn, 0, (size_t)rows * kp * 2, stream);
    k_three_interp<<<(rows + 255) / 256, 256, 0, stream>>>(uxyz, kxyz, kfeat, bufIn,
                                                           nu, nk, ck, kp, rows);
    if (ufeat) {
      int tot = rows * cu;
      k_copy_bf16_cols<<<(tot + 255) / 256, 256, 0, stream>>>(ufeat, bufIn, cu, cu,
                                                              kp, ck, tot);
    } else {  // FP1 skip feats are the raw f32 input features (3 ch)
      int tot = rows * cu;
      k_f32cols_bf16<<<(tot + 255) / 256, 256, 0, stream>>>(feat0, bufIn, cu, kp, ck, tot);
    }
    layer_bf(bufIn,  rows, fp[m][0], fpwb[m][0], kp,                    bufHid, fp_dims[m][1], 0);
    layer_bf(bufHid, rows, fp[m][1], fpwb[m][1], rup(fp[m][1].cin, 32), outb,   fp[m][1].cout, 0);
  };
  run_fp(0, lx[3], lx[4], 64,   16,   saf[4], 1024, saf[3], 1024, fpf[0]);
  run_fp(1, lx[2], lx[3], 256,  64,   fpf[0], 512,  saf[2], 512,  fpf[1]);
  run_fp(2, lx[1], lx[2], 1024, 256,  fpf[1], 512,  saf[1], 256,  fpf[2]);
  run_fp(3, lx[0], lx[1], 8192, 1024, fpf[2], 256,  nullptr, 3,   fpf[3]);

  // -------------------- final linear (+bias cancels) + BN -> d_out --------------------
  gemm(fpf[3], finwb, bufOut, B_ * N_, C_OUT_, 256);
  k_bn_stats<<<C_OUT_, 256, 0, stream>>>(bufOut, mv, B_ * N_, C_OUT_);
  { int tot = B_ * N_ * C_OUT_;
    k_bn_norm_f32<<<(tot + 255) / 256, 256, 0, stream>>>(bufOut, mv, fin.g, fin.b,
                                                         (float*)d_out, C_OUT_, tot); }
}
